// SpectralInitializer_25563645346577
// MI455X (gfx1250) — matrix operand order repaired and checked by reading, not compiled, not run on hardware
//
#include <hip/hip_runtime.h>
#include <hip/hip_fp16.h>

typedef __attribute__((ext_vector_type(16))) _Float16 v16h;
typedef __attribute__((ext_vector_type(8)))  float    v8f;

#define D_DIM 384
#define NB    32

// ---------------- Threefry2x32-20 (JAX-style PRNG skeleton) ----------------
__device__ __forceinline__ void threefry2x32(uint32_t k0, uint32_t k1,
                                             uint32_t c0, uint32_t c1,
                                             uint32_t& o0, uint32_t& o1) {
  const uint32_t R[8] = {13, 15, 26, 6, 17, 29, 16, 24};
  uint32_t ks[3] = {k0, k1, k0 ^ k1 ^ 0x1BD11BDAu};
  uint32_t x0 = c0 + k0, x1 = c1 + k1;
#pragma unroll
  for (int g = 0; g < 5; ++g) {
#pragma unroll
    for (int r = 0; r < 4; ++r) {
      x0 += x1;
      uint32_t rr = R[(g & 1) * 4 + r];
      x1 = (x1 << rr) | (x1 >> (32 - rr));
      x1 ^= x0;
    }
    x0 += ks[(g + 1) % 3];
    x1 += ks[(g + 2) % 3] + (uint32_t)(g + 1);
  }
  o0 = x0; o1 = x1;
}

// ---------------- Kernel 1: 4x4 block-mean pooling to 16x16 grid -----------
// One workgroup per (batch, cell). Streams the 201MB features tensor once.
__global__ void pool16_kernel(const float* __restrict__ feats,
                              float* __restrict__ p16f,
                              _Float16* __restrict__ p16h) {
  int blk  = blockIdx.x;
  int b    = blk >> 8;
  int cell = blk & 255;
  int cy = cell >> 4, cx = cell & 15;
  const float* base = feats + (size_t)b * 4096 * D_DIM;
  for (int d = threadIdx.x; d < D_DIM; d += blockDim.x) {
    float acc = 0.f;
#pragma unroll
    for (int py = 0; py < 4; ++py)
#pragma unroll
      for (int px = 0; px < 4; ++px) {
        int y = cy * 4 + py, x = cx * 4 + px;
        acc += base[(size_t)(y * 64 + x) * D_DIM + d];
      }
    acc *= (1.0f / 16.0f);
    size_t o = ((size_t)b * 256 + cell) * D_DIM + d;
    p16f[o] = acc;
    p16h[o] = (_Float16)acc;
  }
}

// ---------------- Kernel 2: downsample 16x16 -> 8x8 and 16x16 -> 4x4 -------
// Both derived directly from p16 (L2-resident, no dependency chain).
__global__ void downsample_kernel(const float* __restrict__ p16f,
                                  float* __restrict__ p8f, _Float16* __restrict__ p8h,
                                  float* __restrict__ p4f, _Float16* __restrict__ p4h) {
  int blk = blockIdx.x;
  int b = blk / 80;
  int c = blk % 80;
  if (c < 64) {                      // 8x8 cell = mean of 2x2 p16 cells
    int hy = c >> 3, hx = c & 7;
    for (int d = threadIdx.x; d < D_DIM; d += blockDim.x) {
      float acc = 0.f;
#pragma unroll
      for (int dy = 0; dy < 2; ++dy)
#pragma unroll
        for (int dx = 0; dx < 2; ++dx) {
          int g = (hy * 2 + dy) * 16 + (hx * 2 + dx);
          acc += p16f[((size_t)b * 256 + g) * D_DIM + d];
        }
      acc *= 0.25f;
      size_t o = ((size_t)b * 64 + c) * D_DIM + d;
      p8f[o] = acc; p8h[o] = (_Float16)acc;
    }
  } else {                           // 4x4 cell = mean of 4x4 p16 cells
    int q = c - 64;
    int qy = q >> 2, qx = q & 3;
    for (int d = threadIdx.x; d < D_DIM; d += blockDim.x) {
      float acc = 0.f;
#pragma unroll
      for (int dy = 0; dy < 4; ++dy)
#pragma unroll
        for (int dx = 0; dx < 4; ++dx) {
          int g = (qy * 4 + dy) * 16 + (qx * 4 + dx);
          acc += p16f[((size_t)b * 256 + g) * D_DIM + d];
        }
      acc *= (1.0f / 16.0f);
      size_t o = ((size_t)b * 16 + q) * D_DIM + d;
      p4f[o] = acc; p4h[o] = (_Float16)acc;
    }
  }
}

// ---------------- Kernel 3: k-means++ seeding, one workgroup per (scale,b) -
// Distances computed as ||x||^2 - 2 x.c + ||c||^2; the x.c mat-vec runs on
// v_wmma_f32_16x16x32_f16 with the center broadcast across all 16 B columns.
__global__ void __launch_bounds__(256) seed_kernel(
    const float* __restrict__ p4f,  const _Float16* __restrict__ p4h,
    const float* __restrict__ p8f,  const _Float16* __restrict__ p8h,
    const float* __restrict__ p16f, const _Float16* __restrict__ p16h,
    float* __restrict__ out) {
  __shared__ float diag[256];
  __shared__ float minsq[256];
  __shared__ float dots[256];
  __shared__ float score[256];
  __shared__ _Float16 cenH[D_DIM];
  __shared__ int   s_idx;
  __shared__ float s_sum;

  int blk = blockIdx.x;
  int si  = blk >> 5;         // scale index 0..2
  int b   = blk & 31;         // batch
  int s   = (si == 0) ? 4 : ((si == 1) ? 8 : 16);
  int M   = s * s;

  const float*    Xf = (si == 0 ? p4f : (si == 1 ? p8f : p16f)) + (size_t)b * M * D_DIM;
  const _Float16* Xh = (si == 0 ? p4h : (si == 1 ? p8h : p16h)) + (size_t)b * M * D_DIM;
  float* outB = out + ((size_t)b * 12 + (size_t)si * 4) * D_DIM;

  int tid  = threadIdx.x;
  int lane = tid & 31;
  int wave = tid >> 5;

  // ||x_i||^2 in the same (f16) precision as the WMMA dot products.
  for (int i = tid; i < M; i += blockDim.x) {
    float acc = 0.f;
    for (int d = 0; d < D_DIM; ++d) {
      float v = (float)Xh[(size_t)i * D_DIM + d];
      acc += v * v;
    }
    diag[i] = acc;
  }
  __syncthreads();

  // Key derivation (uniform across threads): key(42) -> fold_in(si) ->
  // split(B)[b] -> split(2) -> (k0 for randint, k1 for 3 categorical steps).
  uint32_t bk0, bk1, kb0, kb1, ka0, ka1, kc0, kc1;
  threefry2x32(0u, 42u, 0u, (uint32_t)si, bk0, bk1);
  threefry2x32(bk0, bk1, 2u * (uint32_t)b, 2u * (uint32_t)b + 1u, kb0, kb1);
  threefry2x32(kb0, kb1, 0u, 1u, ka0, ka1);
  threefry2x32(kb0, kb1, 2u, 3u, kc0, kc1);
  uint32_t r0, r1;
  threefry2x32(ka0, ka1, 0u, 0u, r0, r1);
  int idx = (int)(r0 % (uint32_t)M);

  // Emit first center (fp32 source).
  for (int d = tid; d < D_DIM; d += blockDim.x) outB[d] = Xf[(size_t)idx * D_DIM + d];

  int numTiles = M >> 4;   // 1, 4 or 16 row-tiles of 16

  for (int step = 0; step < 4; ++step) {
    // Stage current center (f16) in LDS.
    for (int d = tid; d < D_DIM; d += blockDim.x) cenH[d] = Xh[(size_t)idx * D_DIM + d];
    __syncthreads();

    // dots[i] = x_i . c via WMMA; wave-uniform tile loop keeps EXEC all-1s.
    for (int tile = wave; tile < numTiles; tile += 8) {
      v8f acc = {};
      int row = tile * 16 + (lane & 15);
      const _Float16* xrow = Xh + (size_t)row * D_DIM;
      for (int kb = 0; kb < D_DIM; kb += 32) {
        // A (16x32 f16): lanes 0-15 hold K {0..7,16..23}, lanes 16-31 hold K {8..15,24..31}
        int aoff = (lane < 16) ? 0 : 8;
        v16h A, Bm;
#pragma unroll
        for (int h = 0; h < 8; ++h) {
          A[h]     = xrow[kb + aoff + h];
          A[h + 8] = xrow[kb + aoff + 16 + h];
        }
        // B (32x16 f16), column = center for all N: lanes 0-15 K {0..7,16..23}, lanes 16-31 K {8..15,24..31}
        int boff = (lane < 16) ? 0 : 8;
#pragma unroll
        for (int h = 0; h < 16; ++h) Bm[h] = cenH[kb + boff + (h < 8 ? h : h + 8)];
        acc = __builtin_amdgcn_wmma_f32_16x16x32_f16(
            /*neg_a=*/false, A, /*neg_b=*/false, Bm,
            /*c_mod=*/(short)0, acc, /*reuse_a=*/false, /*reuse_b=*/false);
      }
      // Column N=0 lives in lane 0 (rows 0-7) and lane 16 (rows 8-15).
      if (lane == 0) {
#pragma unroll
        for (int v = 0; v < 8; ++v) dots[tile * 16 + v] = acc[v];
      } else if (lane == 16) {
#pragma unroll
        for (int v = 0; v < 8; ++v) dots[tile * 16 + 8 + v] = acc[v];
      }
    }
    __syncthreads();

    float dcen = diag[idx];
    for (int i = tid; i < M; i += blockDim.x) {
      float d2 = fmaxf(diag[i] - 2.f * dots[i] + dcen, 0.f);
      minsq[i] = (step == 0) ? d2 : fminf(minsq[i], d2);
    }
    __syncthreads();

    if (step == 3) break;

    if (tid == 0) {
      float ssum = 0.f;
      for (int i = 0; i < M; ++i) ssum += minsq[i];
      s_sum = ssum;
    }
    __syncthreads();
    float inv = 1.f / (s_sum + 1e-8f);

    // categorical(step_key) via Gumbel-argmax over M probabilities.
    uint32_t sk0, sk1;
    threefry2x32(kc0, kc1, 2u * (uint32_t)step, 2u * (uint32_t)step + 1u, sk0, sk1);
    for (int i = tid; i < M; i += blockDim.x) {
      uint32_t w0, w1;
      threefry2x32(sk0, sk1, (uint32_t)i, 0u, w0, w1);
      float u = ((float)(w0 >> 8)) * 0x1p-24f + 0x1p-25f;   // (0,1)
      float g = -logf(-logf(u));
      score[i] = logf(minsq[i] * inv + 1e-12f) + g;
    }
    __syncthreads();
    if (tid == 0) {
      int best = 0; float bs = score[0];
      for (int i = 1; i < M; ++i) if (score[i] > bs) { bs = score[i]; best = i; }
      s_idx = best;
    }
    __syncthreads();
    idx = s_idx;

    // Emit this center (fp32 source).
    for (int d = tid; d < D_DIM; d += blockDim.x)
      outB[(size_t)(step + 1) * D_DIM + d] = Xf[(size_t)idx * D_DIM + d];
    __syncthreads();
  }
}

// ---------------- Host-side launcher ---------------------------------------
extern "C" void kernel_launch(void* const* d_in, const int* in_sizes, int n_in,
                              void* d_out, int out_size, void* d_ws, size_t ws_size,
                              hipStream_t stream) {
  (void)in_sizes; (void)n_in; (void)out_size; (void)ws_size;
  const float* feats = (const float*)d_in[0];
  float* out = (float*)d_out;
  char* ws = (char*)d_ws;

  // Workspace layout (bytes):
  float*    p16f = (float*)(ws + 0);                 // 32*256*384*4 = 12,582,912
  float*    p8f  = (float*)(ws + 12582912);          //  3,145,728
  float*    p4f  = (float*)(ws + 15728640);          //    786,432
  _Float16* p16h = (_Float16*)(ws + 16515072);       //  6,291,456
  _Float16* p8h  = (_Float16*)(ws + 22806528);       //  1,572,864
  _Float16* p4h  = (_Float16*)(ws + 24379392);       //    393,216  (end 24,772,608)

  pool16_kernel<<<NB * 256, 128, 0, stream>>>(feats, p16f, p16h);
  downsample_kernel<<<NB * 80, 128, 0, stream>>>(p16f, p8f, p8h, p4f, p4h);
  seed_kernel<<<96, 256, 0, stream>>>(p4f, p4h, p8f, p8h, p16f, p16h, out);
}
